// BackboneMultiviewGmCoPo_45698452029585
// MI455X (gfx1250) — compile-verified, matrix operand order checked by reading, not run on hardware
//
#include <hip/hip_runtime.h>
#include <hip/hip_bf16.h>

// ---------------------------------------------------------------------------
// MI455X (gfx1250) implementation.
// Memory-bound problem (~0.6 GB moved -> ~25-30us @ 23.3 TB/s); GEMM FLOPs are
// negligible, so we use bf16 WMMA (v_wmma_f32_16x16x32_bf16) for the two
// correlation GEMMs, fuse the 4D-upsample epilogue into the fine GEMM so the
// 134MB 'c' volume is written exactly once, and emit 'c_flip' via an LDS tile
// transpose so both store orientations stay coalesced.
// ---------------------------------------------------------------------------

#define BATCH 2
#define CCH   128
#define PC    1024    // 32*32 coarse pixels
#define PF    4096    // 64*64 fine pixels

#define INV_SQRT_C 0.08838834764831845f   // 1/sqrt(128)
#define INV_BETA   50.0f                  // 1/0.02
#define R3163      (31.0f/63.0f)          // align_corners 32->64 scale

typedef __attribute__((ext_vector_type(16))) __bf16 v16bf;
typedef __attribute__((ext_vector_type(8)))  float  v8f;

union FragBF {
  v16bf v;
  uint4 q[2];
};

__device__ __forceinline__ unsigned short f32_to_bf16(float f) {
  unsigned int u = __float_as_uint(f);
  u += 0x7FFFu + ((u >> 16) & 1u);        // round-to-nearest-even
  return (unsigned short)(u >> 16);
}

// ---------------------------------------------------------------------------
// Pack [B][C][P] f32 -> [B][P][C] bf16 (transpose via LDS, coalesced both ways)
// block (32,8), grid (P/32, C/32, B)
// ---------------------------------------------------------------------------
__global__ void pack_bf16_kernel(const float* __restrict__ in,
                                 unsigned short* __restrict__ out,
                                 int C, int P) {
  __shared__ float tile[32][33];
  int b  = blockIdx.z;
  int p0 = blockIdx.x * 32;
  int c0 = blockIdx.y * 32;
  const float* src = in + (size_t)b * C * P;
  unsigned short* dst = out + (size_t)b * P * C;
  int tx = threadIdx.x, ty = threadIdx.y;
  for (int j = 0; j < 32; j += 8)
    tile[ty + j][tx] = src[(size_t)(c0 + ty + j) * P + (p0 + tx)];
  __syncthreads();
  for (int j = 0; j < 32; j += 8)
    dst[(size_t)(p0 + ty + j) * C + (c0 + tx)] = f32_to_bf16(tile[tx][ty + j]);
}

// ---------------------------------------------------------------------------
// trans_features: out[b][s][c][p] = (s?f1f:f0f)[b][c][p]   (float4 copy)
// ---------------------------------------------------------------------------
__global__ void copy_tf_kernel(const float* __restrict__ f0,
                               const float* __restrict__ f1,
                               float* __restrict__ out) {
  const int per = CCH * PF / 4;                 // float4 per (b,s) chunk
  int i = blockIdx.x * 256 + threadIdx.x;       // total BATCH*2*per
  int b = i / (2 * per);
  int r = i - b * 2 * per;
  int s = r / per;
  int o = r - s * per;
  const float4* src = (const float4*)(s ? f1 : f0) + (size_t)b * per + o;
  ((float4*)out)[i] = *src;
}

// ---------------------------------------------------------------------------
// trans_features_coarse: bilinear 32->64 (align_corners) per channel
// ---------------------------------------------------------------------------
__global__ void upfeat_kernel(const float* __restrict__ f0c,
                              const float* __restrict__ f1c,
                              float* __restrict__ out) {
  int i = blockIdx.x * 256 + threadIdx.x;       // total 2^21
  int ox = i & 63;
  int oy = (i >> 6) & 63;
  int c  = (i >> 12) & 127;
  int s  = (i >> 19) & 1;
  int b  = i >> 20;
  float syf = oy * R3163; int y0 = (int)syf; float wy = syf - y0; int y1 = min(y0 + 1, 31);
  float sxf = ox * R3163; int x0 = (int)sxf; float wx = sxf - x0; int x1 = min(x0 + 1, 31);
  const float* src = (s ? f1c : f0c) + ((size_t)b * CCH + c) * PC;
  float v = (1.f - wy) * ((1.f - wx) * src[y0 * 32 + x0] + wx * src[y0 * 32 + x1])
          +        wy  * ((1.f - wx) * src[y1 * 32 + x0] + wx * src[y1 * 32 + x1]);
  out[i] = v;
}

// ---------------------------------------------------------------------------
// Coarse correlation GEMM: corrC[b][p][q] = (F0c[p,:].F1c[q,:]) / sqrt(C)
// 64x64 macro tile, 4 waves (2x2), each wave 2x2 wmma 16x16x32 bf16 tiles.
// ---------------------------------------------------------------------------
__global__ void gemm_coarse_kernel(const unsigned short* __restrict__ A,
                                   const unsigned short* __restrict__ Bm,
                                   float* __restrict__ out) {
  int b    = blockIdx.z;
  int pt   = blockIdx.y * 64;
  int qt   = blockIdx.x * 64;
  int lane = threadIdx.x & 31;
  int wid  = threadIdx.x >> 5;
  int wm   = (wid >> 1) * 32;
  int wn   = (wid & 1) * 32;
  int l15  = lane & 15;
  int half = lane >> 4;
  const unsigned short* Ab = A  + (size_t)b * PC * CCH;
  const unsigned short* Bb = Bm + (size_t)b * PC * CCH;

  v8f acc[2][2] = {};
  for (int k = 0; k < CCH; k += 32) {
    FragBF a[2], bb[2];
    for (int t = 0; t < 2; t++) {
      const unsigned short* ap = Ab + (size_t)(pt + wm + t * 16 + l15) * CCH + k + half * 8;
      a[t].q[0] = *(const uint4*)ap;
      a[t].q[1] = *(const uint4*)(ap + 16);
      const unsigned short* bp = Bb + (size_t)(qt + wn + t * 16 + l15) * CCH + k + half * 8;
      bb[t].q[0] = *(const uint4*)bp;
      bb[t].q[1] = *(const uint4*)(bp + 16);
    }
    for (int tm = 0; tm < 2; tm++)
      for (int tn = 0; tn < 2; tn++)
        acc[tm][tn] = __builtin_amdgcn_wmma_f32_16x16x32_bf16(
            false, a[tm].v, false, bb[tn].v, (short)0, acc[tm][tn], false, false);
  }
  float* ob = out + (size_t)b * PC * PC;
  for (int tm = 0; tm < 2; tm++)
    for (int tn = 0; tn < 2; tn++)
      for (int r = 0; r < 8; r++) {
        int m = pt + wm + tm * 16 + r + 8 * half;
        int n = qt + wn + tn * 16 + l15;
        ob[(size_t)m * PC + n] = acc[tm][tn][r] * INV_SQRT_C;
      }
}

// ---------------------------------------------------------------------------
// Separable 4D upsample, pass 1: interpolate over the (u,v) pair.
// tmp[b][pc][q] = bilerp_{u,v}(corrC[b][pc][:,:], q=(y1,x1))
// ---------------------------------------------------------------------------
__global__ void ups_uv_kernel(const float* __restrict__ corrc,
                              float* __restrict__ tmp) {
  int i  = blockIdx.x * 256 + threadIdx.x;      // total B*PC*PF = 2^23
  int q  = i & 4095;
  int pc = (i >> 12) & 1023;
  int b  = i >> 22;
  int x1 = q & 63, y1 = q >> 6;
  float su = y1 * R3163; int u0 = (int)su; float wu = su - u0; int u1 = min(u0 + 1, 31);
  float sv = x1 * R3163; int v0 = (int)sv; float wv = sv - v0; int v1 = min(v0 + 1, 31);
  const float* src = corrc + ((size_t)b * PC + pc) * PC;
  float val = (1.f - wu) * ((1.f - wv) * src[u0 * 32 + v0] + wv * src[u0 * 32 + v1])
            +        wu  * ((1.f - wv) * src[u1 * 32 + v0] + wv * src[u1 * 32 + v1]);
  tmp[i] = val;
}

// ---------------------------------------------------------------------------
// Fused fine GEMM + (h,w)-interp of coarse volume + 0.5*(..) epilogue.
// Writes c[p0][q] directly, and c_flip[q][p0] via LDS tile transpose.
// Tile = 64 p0 (fixed y0 = blockIdx.y) x 64 q (fixed y1 = blockIdx.x).
// ---------------------------------------------------------------------------
__global__ void fused_fine_kernel(const unsigned short* __restrict__ A,
                                  const unsigned short* __restrict__ Bm,
                                  const float* __restrict__ tmp,
                                  float* __restrict__ cOut,
                                  float* __restrict__ cFlip) {
  __shared__ float lds[64 * 65];
  int b    = blockIdx.z;
  int pt   = blockIdx.y * 64;
  int qt   = blockIdx.x * 64;
  int lane = threadIdx.x & 31;
  int wid  = threadIdx.x >> 5;
  int wm   = (wid >> 1) * 32;
  int wn   = (wid & 1) * 32;
  int l15  = lane & 15;
  int half = lane >> 4;
  const unsigned short* Ab = A  + (size_t)b * PF * CCH;
  const unsigned short* Bb = Bm + (size_t)b * PF * CCH;

  v8f acc[2][2] = {};
  for (int k = 0; k < CCH; k += 32) {
    FragBF a[2], bb[2];
    for (int t = 0; t < 2; t++) {
      const unsigned short* ap = Ab + (size_t)(pt + wm + t * 16 + l15) * CCH + k + half * 8;
      a[t].q[0] = *(const uint4*)ap;
      a[t].q[1] = *(const uint4*)(ap + 16);
      const unsigned short* bp = Bb + (size_t)(qt + wn + t * 16 + l15) * CCH + k + half * 8;
      bb[t].q[0] = *(const uint4*)bp;
      bb[t].q[1] = *(const uint4*)(bp + 16);
    }
    for (int tm = 0; tm < 2; tm++)
      for (int tn = 0; tn < 2; tn++)
        acc[tm][tn] = __builtin_amdgcn_wmma_f32_16x16x32_bf16(
            false, a[tm].v, false, bb[tn].v, (short)0, acc[tm][tn], false, false);
  }

  // (h,w)-pair interpolation weights: y0 = blockIdx.y is tile-uniform.
  float sy = blockIdx.y * R3163;
  int hy0 = (int)sy;  float ay = sy - hy0;  int hy1 = min(hy0 + 1, 31);
  const float* tb = tmp + (size_t)b * PC * PF;
  float* co = cOut  + (size_t)b * PF * PF;
  float* cf = cFlip + (size_t)b * PF * PF;

  for (int tm = 0; tm < 2; tm++)
    for (int tn = 0; tn < 2; tn++)
      for (int r = 0; r < 8; r++) {
        int ml = wm + tm * 16 + r + 8 * half;       // local m == x0
        int nl = wn + tn * 16 + l15;                // local n == x1
        float sx = ml * R3163;
        int hx0 = (int)sx;  float ax = sx - hx0;  int hx1 = min(hx0 + 1, 31);
        int q = qt + nl;
        float t00 = tb[(size_t)(hy0 * 32 + hx0) * PF + q];
        float t01 = tb[(size_t)(hy0 * 32 + hx1) * PF + q];
        float t10 = tb[(size_t)(hy1 * 32 + hx0) * PF + q];
        float t11 = tb[(size_t)(hy1 * 32 + hx1) * PF + q];
        float up  = (1.f - ay) * ((1.f - ax) * t00 + ax * t01)
                  +        ay  * ((1.f - ax) * t10 + ax * t11);
        float val = 0.5f * (acc[tm][tn][r] * INV_SQRT_C + up);
        co[(size_t)(pt + ml) * PF + q] = val;
        lds[ml * 65 + nl] = val;
      }
  __syncthreads();
  // coalesced transposed stores: consecutive threads -> consecutive p0
  for (int it = 0; it < 32; it++) {
    int flat = it * 128 + threadIdx.x;
    int nl = flat >> 6;
    int ml = flat & 63;
    cf[(size_t)(qt + nl) * PF + (pt + ml)] = lds[ml * 65 + nl];
  }
}

// ---------------------------------------------------------------------------
// Row softmax (temperature beta) + soft-argmax -> pixel flow.
// One 256-thread block per row of 4096; single read, values kept in registers.
// ---------------------------------------------------------------------------
__device__ __forceinline__ float block_reduce_add(float v, float* red) {
  int t = threadIdx.x;
  red[t] = v; __syncthreads();
  for (int s = 128; s > 0; s >>= 1) { if (t < s) red[t] += red[t + s]; __syncthreads(); }
  float r = red[0]; __syncthreads();
  return r;
}

__global__ void softargmax_kernel(const float* __restrict__ cmat,
                                  float* __restrict__ flow) {
  __shared__ float red[256];
  int b   = blockIdx.y;
  int row = blockIdx.x;
  int t   = threadIdx.x;
  const float* r = cmat + ((size_t)b * PF + row) * PF;
  const float4* r4 = (const float4*)(r + t * 16);
  float4 q0 = r4[0], q1 = r4[1], q2 = r4[2], q3 = r4[3];
  float v[16] = { q0.x, q0.y, q0.z, q0.w, q1.x, q1.y, q1.z, q1.w,
                  q2.x, q2.y, q2.z, q2.w, q3.x, q3.y, q3.z, q3.w };
  float mx = -3.4e38f;
  for (int i = 0; i < 16; i++) mx = fmaxf(mx, v[i]);
  // block max
  red[t] = mx; __syncthreads();
  for (int s = 128; s > 0; s >>= 1) { if (t < s) red[t] = fmaxf(red[t], red[t + s]); __syncthreads(); }
  mx = red[0]; __syncthreads();

  float s = 0.f, sx = 0.f, sy = 0.f;
  for (int i = 0; i < 16; i++) {
    int q = t * 16 + i;
    float e = __expf((v[i] - mx) * INV_BETA);
    s  += e;
    sx += e * ((q & 63) * (2.0f / 63.0f) - 1.0f);
    sy += e * ((q >> 6) * (2.0f / 63.0f) - 1.0f);
  }
  float S  = block_reduce_add(s,  red);
  float SX = block_reduce_add(sx, red);
  float SY = block_reduce_add(sy, red);
  if (t == 0) {
    float gx = SX / S, gy = SY / S;
    float mxp = (gx + 1.f) * 31.5f;
    float myp = (gy + 1.f) * 31.5f;
    int tx = row & 63, ty = row >> 6;
    flow[(size_t)b * 2 * PF + row]      = mxp - (float)tx;
    flow[(size_t)b * 2 * PF + PF + row] = myp - (float)ty;
  }
}

// ---------------------------------------------------------------------------
extern "C" void kernel_launch(void* const* d_in, const int* in_sizes, int n_in,
                              void* d_out, int out_size, void* d_ws, size_t ws_size,
                              hipStream_t stream) {
  const float* f0c = (const float*)d_in[0];
  const float* f1c = (const float*)d_in[1];
  const float* f0f = (const float*)d_in[2];
  const float* f1f = (const float*)d_in[3];

  float* out      = (float*)d_out;
  float* out_tf   = out;                       // [2,2,128,64,64]   2,097,152
  float* out_tfc  = out + 2097152;             // [2,2,128,64,64]   2,097,152
  float* out_c    = out + 4194304;             // [2,1,64,64,64,64] 33,554,432
  float* out_cfl  = out + 37748736;            // [2,1,64,64,64,64] 33,554,432
  float* out_flow = out + 71303168;            // [2,2,64,64]       16,384
  float* out_flf  = out + 71319552;            // [2,2,64,64]       16,384

  // workspace carve-up (~45 MB)
  unsigned short* Af = (unsigned short*)d_ws;                  // [B][PF][C] bf16
  unsigned short* Bf = Af + (size_t)BATCH * PF * CCH;
  unsigned short* Ac = Bf + (size_t)BATCH * PF * CCH;          // [B][PC][C]
  unsigned short* Bc = Ac + (size_t)BATCH * PC * CCH;
  float* corrC = (float*)(Bc + (size_t)BATCH * PC * CCH);      // [B][PC][PC]
  float* tmp   = corrC + (size_t)BATCH * PC * PC;              // [B][PC][PF]

  dim3 blk328(32, 8);
  pack_bf16_kernel<<<dim3(PF / 32, CCH / 32, BATCH), blk328, 0, stream>>>(f0f, Af, CCH, PF);
  pack_bf16_kernel<<<dim3(PF / 32, CCH / 32, BATCH), blk328, 0, stream>>>(f1f, Bf, CCH, PF);
  pack_bf16_kernel<<<dim3(PC / 32, CCH / 32, BATCH), blk328, 0, stream>>>(f0c, Ac, CCH, PC);
  pack_bf16_kernel<<<dim3(PC / 32, CCH / 32, BATCH), blk328, 0, stream>>>(f1c, Bc, CCH, PC);

  copy_tf_kernel<<<2048, 256, 0, stream>>>(f0f, f1f, out_tf);
  upfeat_kernel<<<(1 << 21) / 256, 256, 0, stream>>>(f0c, f1c, out_tfc);

  gemm_coarse_kernel<<<dim3(16, 16, BATCH), 128, 0, stream>>>(Ac, Bc, corrC);
  ups_uv_kernel<<<(1 << 23) / 256, 256, 0, stream>>>(corrC, tmp);
  fused_fine_kernel<<<dim3(64, 64, BATCH), 128, 0, stream>>>(Af, Bf, tmp, out_c, out_cfl);

  softargmax_kernel<<<dim3(PF, BATCH), 256, 0, stream>>>(out_c,   out_flow);
  softargmax_kernel<<<dim3(PF, BATCH), 256, 0, stream>>>(out_cfl, out_flf);
}
